// MDR_InteractingLayer_38104949850309
// MI455X (gfx1250) — compile-verified, matrix operand order checked by reading, not run on hardware
//
#include <hip/hip_runtime.h>
#include <hip/hip_bf16.h>

typedef __attribute__((ext_vector_type(16))) _Float16 v16h;
typedef __attribute__((ext_vector_type(2)))  __fp16   pkh2;   // cvt_pkrtz result type
typedef __attribute__((ext_vector_type(8)))  float    v8f;

#define B_  8192
#define F_  40
#define E_  128
#define H_  4
#define DH_ 32
#define FP_ 48      // F padded to multiple of 16

// LDS strides (element units)
#define XSTR 136    // f16, X rows
#define WSTR 136    // f16, transposed weight rows
#define QSTR 136    // f16, Q/K rows
#define SSTR 52     // f32, score rows
#define PSTR 64     // f16, attn-prob rows (k-dim padded to 64)
#define VSTR 72     // f16, V-transposed rows (j-dim padded)

// LDS byte offsets (phase-aliased)
#define OFF_X   0
#define OFF_W   13056                    // 48*136*2
#define OFF_S   0                        // aliases X+W after projections
#define OFF_Q   47872                    // 13056 + 128*136*2
#define OFF_K   60928
#define OFF_P   47872                    // aliases Q+K after scores
#define OFF_VT  73984
#define OFF_R   92416
#define SMEM_BYTES 116992                // + 48*128*4

__device__ __forceinline__ v8f wmma_f16(v16h a, v16h b, v8f c) {
  return __builtin_amdgcn_wmma_f32_16x16x32_f16(false, a, false, b, (short)0, c,
                                                false, false);
}

// Fragment loader for a row-major [row][k] f16 LDS tile.
// A-matrix: row = m (lane%16), per documented 16x32 f16 layout.
// B-matrix: mirrored layout with row = n.
__device__ __forceinline__ v16h load_frag(const _Float16* base, int stride,
                                          int rowBase, int kBase) {
  const int lane = threadIdx.x & 31;
  const int r  = rowBase + (lane & 15);
  const int kh = lane >> 4;
  v16h f;
#pragma unroll
  for (int v = 0; v < 8; ++v) {
    const int k = kBase + ((v >> 2) << 4) + (kh << 3) + ((v & 3) << 1);
    const _Float16* p = base + r * stride + k;
    f[2 * v]     = p[0];
    f[2 * v + 1] = p[1];
  }
  return f;
}

// One projection GEMM: Out(48x128) = X(48x128) * W(128x128), store per STEP:
//   STEP 0 -> sK (f16), 1 -> sQ (f16), 2 -> sVt (f16, transposed), 3 -> sR (f32)
template <int STEP>
__device__ __forceinline__ void proj_gemm(const _Float16* sX, const _Float16* sWt,
                                          _Float16* sQ, _Float16* sK,
                                          _Float16* sVt, float* sR,
                                          int wave, int lane) {
  const int nt0 = wave * 2;
  // B fragments depend only on (nt, ks): load once, reuse across 3 m-tiles.
  v16h bf[2][4];
#pragma unroll
  for (int ks = 0; ks < 4; ++ks) {
    bf[0][ks] = load_frag(sWt, WSTR, (nt0 + 0) * 16, ks * 32);
    bf[1][ks] = load_frag(sWt, WSTR, (nt0 + 1) * 16, ks * 32);
  }
  const int ncol = lane & 15;
  const int e0 = (nt0 + 0) * 16 + ncol;
  const int e1 = (nt0 + 1) * 16 + ncol;
#pragma unroll
  for (int mt = 0; mt < 3; ++mt) {
    v8f acc0 = {}; v8f acc1 = {};
#pragma unroll
    for (int ks = 0; ks < 4; ++ks) {
      v16h a = load_frag(sX, XSTR, mt * 16, ks * 32);
      acc0 = wmma_f16(a, bf[0][ks], acc0);
      acc1 = wmma_f16(a, bf[1][ks], acc1);
    }
    const int mrow = mt * 16 + ((lane >> 4) << 3);
    if constexpr (STEP == 2) {
      // lane owns 8 consecutive m for fixed e: pack -> two 128-bit LDS stores
      union { pkh2 h2[4]; uint4 u; } p0, p1;
#pragma unroll
      for (int rr = 0; rr < 4; ++rr) {
        p0.h2[rr] = __builtin_amdgcn_cvt_pkrtz(acc0[2 * rr], acc0[2 * rr + 1]);
        p1.h2[rr] = __builtin_amdgcn_cvt_pkrtz(acc1[2 * rr], acc1[2 * rr + 1]);
      }
      *(uint4*)(sVt + e0 * VSTR + mrow) = p0.u;
      *(uint4*)(sVt + e1 * VSTR + mrow) = p1.u;
    } else {
#pragma unroll
      for (int r = 0; r < 8; ++r) {
        const int m = mrow + r;
        if constexpr (STEP == 0) {
          union { pkh2 p; _Float16 h[2]; } cv;
          cv.p = __builtin_amdgcn_cvt_pkrtz(acc0[r], acc1[r]);
          sK[m * QSTR + e0] = cv.h[0];
          sK[m * QSTR + e1] = cv.h[1];
        } else if constexpr (STEP == 1) {
          union { pkh2 p; _Float16 h[2]; } cv;
          cv.p = __builtin_amdgcn_cvt_pkrtz(acc0[r], acc1[r]);
          sQ[m * QSTR + e0] = cv.h[0];
          sQ[m * QSTR + e1] = cv.h[1];
        } else {
          sR[m * E_ + e0] = acc0[r];
          sR[m * E_ + e1] = acc1[r];
        }
      }
    }
  }
}

// Weight matrices are PRE-TRANSPOSED ([n][k] row-major) by the convert kernel,
// so staging is a straight 128-bit copy into the padded-stride LDS tile.
__device__ __forceinline__ void stage_weight(const _Float16* __restrict__ wsrcT,
                                             _Float16* sWt, int tid) {
  __syncthreads();  // previous compute done before restaging sWt
  const uint4* src = (const uint4*)wsrcT;          // 2048 x 16B
  for (int idx = tid; idx < 2048; idx += 128) {
    const int n = idx >> 4, c = idx & 15;
    *(uint4*)(sWt + n * WSTR + c * 8) = src[idx];
  }
  __syncthreads();
}

__global__ void MDR_convert_weights(const float* __restrict__ wq,
                                    const float* __restrict__ wv,
                                    const float* __restrict__ wr,
                                    _Float16* __restrict__ wq16t,
                                    _Float16* __restrict__ wv16t,
                                    _Float16* __restrict__ wr16t) {
  const int i = blockIdx.x * 256 + threadIdx.x;   // 65536 threads
  if (i < 4 * E_ * E_) {                          // transposed: [d][n][k]
    const int dd = i >> 14, rem = i & 16383;
    const int n = rem >> 7, k = rem & 127;
    wq16t[i] = (_Float16)wq[dd * E_ * E_ + k * E_ + n];
  }
  if (i < E_ * E_) {
    const int n = i >> 7, k = i & 127;
    wv16t[i] = (_Float16)wv[k * E_ + n];
    wr16t[i] = (_Float16)wr[k * E_ + n];
  }
}

__global__ __launch_bounds__(128)
void MDR_InteractingLayer_fused(const float* __restrict__ x,
                                const int* __restrict__ dom,
                                const _Float16* __restrict__ wq16t,
                                const _Float16* __restrict__ wv16t,
                                const _Float16* __restrict__ wr16t,
                                const float* __restrict__ gamma,
                                const float* __restrict__ beta,
                                float* __restrict__ out) {
  __shared__ __align__(16) unsigned char smem[SMEM_BYTES];
  _Float16* sX  = (_Float16*)(smem + OFF_X);
  _Float16* sWt = (_Float16*)(smem + OFF_W);
  float*    sS  = (float*)   (smem + OFF_S);
  _Float16* sQ  = (_Float16*)(smem + OFF_Q);
  _Float16* sK  = (_Float16*)(smem + OFF_K);
  _Float16* sP  = (_Float16*)(smem + OFF_P);
  _Float16* sVt = (_Float16*)(smem + OFF_VT);
  float*    sR  = (float*)   (smem + OFF_R);

  const int b    = blockIdx.x;
  const int tid  = threadIdx.x;
  const int wave = tid >> 5;
  const int lane = tid & 31;
  const int d    = dom[b] - 1;

  const _Float16* wqd = wq16t + d * E_ * E_;
  __builtin_prefetch(wq16t, 0, 3);
  __builtin_prefetch(wqd, 0, 3);
  __builtin_prefetch(wv16t, 0, 3);
  __builtin_prefetch(wr16t, 0, 3);

  // ---- stage X (f32 -> f16, vectorized), zero pad rows; zero V^T ----
  const float4* xb4 = (const float4*)(x + (size_t)b * F_ * E_);
  for (int idx = tid; idx < FP_ * (E_ / 4); idx += 128) {
    const int f = idx >> 5, c = idx & 31;
    float4 xv = make_float4(0.f, 0.f, 0.f, 0.f);
    if (f < F_) xv = xb4[f * 32 + c];
    union { _Float16 h[4]; uint2 u; } p;
    p.h[0] = (_Float16)xv.x; p.h[1] = (_Float16)xv.y;
    p.h[2] = (_Float16)xv.z; p.h[3] = (_Float16)xv.w;
    *(uint2*)(sX + f * XSTR + c * 4) = p.u;
  }
  for (int idx = tid; idx < E_ * VSTR / 8; idx += 128) {
    ((uint4*)sVt)[idx] = make_uint4(0u, 0u, 0u, 0u);
  }

  // ---- four projection GEMMs: K, Q(domain), V(transposed), Residual ----
  stage_weight(wq16t, sWt, tid);
  proj_gemm<0>(sX, sWt, sQ, sK, sVt, sR, wave, lane);
  stage_weight(wqd, sWt, tid);
  proj_gemm<1>(sX, sWt, sQ, sK, sVt, sR, wave, lane);
  stage_weight(wv16t, sWt, tid);
  proj_gemm<2>(sX, sWt, sQ, sK, sVt, sR, wave, lane);
  stage_weight(wr16t, sWt, tid);
  proj_gemm<3>(sX, sWt, sQ, sK, sVt, sR, wave, lane);
  __syncthreads();

  // ---- scores: S_h = Q_h K_h^T / sqrt(DH); one head per wave ----
  {
    const int h = wave;
    const _Float16* Qh = sQ + h * DH_;
    const _Float16* Kh = sK + h * DH_;
    float* Sh = sS + h * FP_ * SSTR;
#pragma unroll
    for (int mt = 0; mt < 3; ++mt) {
      v16h a = load_frag(Qh, QSTR, mt * 16, 0);
#pragma unroll
      for (int nt = 0; nt < 3; ++nt) {
        v16h bb = load_frag(Kh, QSTR, nt * 16, 0);
        v8f acc = {};
        acc = wmma_f16(a, bb, acc);
        const int irow = mt * 16 + ((lane >> 4) << 3);
        const int jcol = nt * 16 + (lane & 15);
#pragma unroll
        for (int r = 0; r < 8; ++r)
          Sh[(irow + r) * SSTR + jcol] = acc[r] * 0.17677669529663687f;
      }
    }
  }
  __syncthreads();

  // ---- softmax over j<40 (masked), write f16 probs (padded to 64) ----
  for (int slot = tid; slot < H_ * FP_; slot += 128) {
    const int h = slot / FP_, i = slot % FP_;
    float* Srow = sS + (h * FP_ + i) * SSTR;
    _Float16* Prow = sP + (h * FP_ + i) * PSTR;
    if (i < F_) {
      float mx = -1e30f;
      for (int j = 0; j < F_; ++j) mx = fmaxf(mx, Srow[j]);
      float sum = 0.f;
      for (int j = 0; j < F_; ++j) {
        const float e = __expf(Srow[j] - mx);
        Srow[j] = e;            // cache exp in place (row is private to this thread)
        sum += e;
      }
      const float inv = 1.f / sum;
      for (int j = 0; j < F_; ++j) Prow[j] = (_Float16)(Srow[j] * inv);
      for (int j = F_; j < PSTR; ++j) Prow[j] = (_Float16)0.f;
    } else {
      for (int j = 0; j < PSTR; ++j) Prow[j] = (_Float16)0.f;
    }
  }
  __syncthreads();

  // ---- out = P_h @ V_h, add residual, relu (in place in sR) ----
  {
    const int h = wave;
    const _Float16* Ph  = sP + h * FP_ * PSTR;
    const _Float16* Vth = sVt + (h * DH_) * VSTR;
#pragma unroll
    for (int mt = 0; mt < 3; ++mt) {
#pragma unroll
      for (int nt = 0; nt < 2; ++nt) {
        v8f acc = {};
#pragma unroll
        for (int ks = 0; ks < 2; ++ks) {
          v16h a  = load_frag(Ph,  PSTR, mt * 16, ks * 32);
          v16h bb = load_frag(Vth, VSTR, nt * 16, ks * 32);
          acc = wmma_f16(a, bb, acc);
        }
        const int mrow = mt * 16 + ((lane >> 4) << 3);
        const int e    = h * DH_ + nt * 16 + (lane & 15);
#pragma unroll
        for (int r = 0; r < 8; ++r) {
          const int m = mrow + r;
          if (m < F_) {
            const float v = acc[r] + sR[m * E_ + e];
            sR[m * E_ + e] = fmaxf(v, 0.f);
          }
        }
      }
    }
  }
  __syncthreads();

  // ---- per-domain LayerNorm (eps=1e-6), wave-parallel rows ----
  const float* g = gamma + d * E_;
  const float* be = beta + d * E_;
  float* ob = out + (size_t)b * F_ * E_;
  for (int m = wave; m < F_; m += 4) {
    float xv[4], s = 0.f, ss = 0.f;
#pragma unroll
    for (int q = 0; q < 4; ++q) {
      xv[q] = sR[m * E_ + lane + q * 32];
      s += xv[q];
      ss += xv[q] * xv[q];
    }
#pragma unroll
    for (int off = 16; off > 0; off >>= 1) {
      s  += __shfl_xor(s, off, 32);
      ss += __shfl_xor(ss, off, 32);
    }
    const float mean = s * (1.f / 128.f);
    const float var  = ss * (1.f / 128.f) - mean * mean;
    const float rstd = rsqrtf(var + 1e-6f);
#pragma unroll
    for (int q = 0; q < 4; ++q) {
      const int e = lane + q * 32;
      ob[m * E_ + e] = (xv[q] - mean) * rstd * g[e] + be[e];
    }
  }
}

extern "C" void kernel_launch(void* const* d_in, const int* in_sizes, int n_in,
                              void* d_out, int out_size, void* d_ws, size_t ws_size,
                              hipStream_t stream) {
  const float* x     = (const float*)d_in[0];
  const int*   dom   = (const int*)d_in[1];
  const float* wq    = (const float*)d_in[2];
  const float* wv    = (const float*)d_in[3];
  const float* wr    = (const float*)d_in[4];
  const float* gamma = (const float*)d_in[5];
  const float* beta  = (const float*)d_in[6];
  float* out = (float*)d_out;

  _Float16* wq16t = (_Float16*)d_ws;                          // 4*128*128 halves
  _Float16* wv16t = (_Float16*)((char*)d_ws + 131072);        // 128*128 halves
  _Float16* wr16t = (_Float16*)((char*)d_ws + 163840);        // 128*128 halves

  MDR_convert_weights<<<256, 256, 0, stream>>>(wq, wv, wr, wq16t, wv16t, wr16t);
  MDR_InteractingLayer_fused<<<B_, 128, 0, stream>>>(x, dom, wq16t, wv16t, wr16t,
                                                     gamma, beta, out);
}
